// LightweightDGCNN_22024592294549
// MI455X (gfx1250) — matrix operand order repaired and checked by reading, not compiled, not run on hardware
//
#include <hip/hip_runtime.h>
#include <hip/hip_bf16.h>

// ---------------------------------------------------------------------------
// Problem constants (match reference)
// ---------------------------------------------------------------------------
#define N_NODES  100000
#define N_EDGES  500000
#define N_GRAPHS 128
#define HID      128
#define MSG_K    257          // 2*HID + 1
#define KPAD     288          // MSG_K padded to multiple of 32
#define NKT1     (KPAD / 32)  // 9 K-tiles for GEMM1
#define NKT2     (HID / 32)   // 4 K-tiles for GEMM2
#define TILE_E   64           // edges per block tile
#define EDGE_BLOCKS 1024      // persistent grid for edge conv

// ---------------------------------------------------------------------------
// WMMA types / helpers (gfx1250, wave32)
// ---------------------------------------------------------------------------
typedef __attribute__((ext_vector_type(16))) __bf16        v16bf;
typedef __attribute__((ext_vector_type(8)))  float         v8f;
typedef __attribute__((ext_vector_type(4)))  unsigned int  v4u;

union FragU { v16bf b; v4u q[2]; };

// RNE conversion: used only for one-time weight staging.
__device__ __forceinline__ unsigned short f2bf(float f) {
    unsigned int u = __float_as_uint(f);
    unsigned int r = u + 0x7FFFu + ((u >> 16) & 1u);
    return (unsigned short)(r >> 16);
}
// Truncating conversion: hot per-tile paths (1 VALU op).
__device__ __forceinline__ unsigned short f2bf_rz(float f) {
    return (unsigned short)(__float_as_uint(f) >> 16);
}

// ---------------------------------------------------------------------------
// Monotonic float<->u32 key: a <= b  <=>  f32key(a) <= f32key(b) (unsigned).
// Lets segment-max use a single branchless global_atomic_max_u32.
// Key 0 (buffer init) decodes to NaN -> caught by the isfinite->0 finalize.
// ---------------------------------------------------------------------------
__device__ __forceinline__ unsigned int f32key(float v) {
    const unsigned int u = __float_as_uint(v);
    return u ^ (0x80000000u | (unsigned int)((int)u >> 31));
}
__device__ __forceinline__ float keyf32(unsigned int k) {
    const unsigned int x = 0x80000000u | ~(unsigned int)((int)k >> 31);
    return __uint_as_float(k ^ x);
}

// ---------------------------------------------------------------------------
// Fragment-native LDS layouts (ISA 7.12.2).  Each 16x32 (A) / 32x16 (B) bf16
// fragment is 512 contiguous bf16; lane L's 16 values at frag*512 + L*16,
// VGPR order v = 2*j + hi.  Fragment load = two ds_load_b128 per lane.
// ---------------------------------------------------------------------------
__device__ __forceinline__ int a_slot(int m, int k, int nKT) {
    const int mt = m >> 4, ml = m & 15;
    const int kt = k >> 5, kl = k & 31;
    const int pb = kl & ~1, hi = kl & 1;
    const int hL = (pb >> 3) & 1;
    const int j  = ((pb & 7) >> 1) + ((pb >> 4) << 2);
    return ((mt * nKT + kt) << 9) + ((ml + (hL << 4)) << 4) + (j << 1) + hi;
}

__device__ __forceinline__ int b_slot(int k, int n, int nKT) {
    const int nt = n >> 4, nl = n & 15;
    const int kt = k >> 5, kl = k & 31;
    const int hL = (kl >> 3) & 1, rem = (kl & 7) + ((kl >> 4) << 3);
    return ((nt * nKT + kt) << 9) + ((nl + (hL << 4)) << 4) + rem;
}

__device__ __forceinline__ v16bf load_frag(const unsigned short* base, int fragId, int lane) {
    const v4u* p = (const v4u*)(base + (fragId << 9) + (lane << 4));
    FragU f;
    f.q[0] = p[0];
    f.q[1] = p[1];
    return f.b;
}

// ---------------------------------------------------------------------------
// Kernel: fill a u32 buffer with a pattern
// ---------------------------------------------------------------------------
__global__ void fill_u32_kernel(unsigned int* p, long n, unsigned int v) {
    long i = (long)blockIdx.x * blockDim.x + threadIdx.x;
    long stride = (long)gridDim.x * blockDim.x;
    for (; i < n; i += stride) p[i] = v;
}

// ---------------------------------------------------------------------------
// Kernel: node encoder  h = relu([feat | key*W_enc + b_enc] @ W0 + b0)
// ---------------------------------------------------------------------------
__global__ void encoder_kernel(const float* __restrict__ x,
                               const float* __restrict__ W_enc,
                               const float* __restrict__ b_enc,
                               const float* __restrict__ W0,
                               const float* __restrict__ b0,
                               float* __restrict__ h) {
    __shared__ float sx[16];
    __shared__ float sin[47];
    const int n = blockIdx.x;
    const int t = threadIdx.x;          // 128
    if (t < 16) sx[t] = x[(size_t)n * 16 + t];
    __syncthreads();
    if (t < 15)      sin[t] = sx[1 + t];
    else if (t < 47) { int j = t - 15; sin[t] = sx[0] * W_enc[j] + b_enc[j]; }
    __syncthreads();
    float a = b0[t];
#pragma unroll 1
    for (int k = 0; k < 47; ++k) a += sin[k] * W0[k * HID + t];
    h[(size_t)n * HID + t] = a > 0.f ? a : 0.f;
}

// ---------------------------------------------------------------------------
// Kernel: one EdgeConv layer message MLP + segment-max scatter.
//   msg  = [h[src]-h[dst] | h[dst] | key[src]-key[dst] | pad]   (64 x 288)
//   mid  = leaky_relu(msg @ Wa + ba)                            (64 x 128)
//   out  = mid @ Wb + bb  --atomic-max(key)--> agg[dst]         (64 x 128)
// 256 threads = 8 waves; wave w owns M-tile (w&3) x N-tiles 4*(w>>2)..+3.
// ---------------------------------------------------------------------------
__global__ void edge_conv_kernel(const float* __restrict__ hIn, long hStride,
                                 const float* __restrict__ x,      // key = x[:,0]
                                 const int* __restrict__ src,
                                 const int* __restrict__ dst,
                                 int nEdges,
                                 const float* __restrict__ Wa,     // [257][128]
                                 const float* __restrict__ ba,
                                 const float* __restrict__ Wb,     // [128][128]
                                 const float* __restrict__ bb,
                                 unsigned int* __restrict__ agg)   // [N][128] keys, 0 init
{
    extern __shared__ unsigned char smemRaw[];
    unsigned short* sWaF  = (unsigned short*)smemRaw;              // 8*9  frags = 36864
    unsigned short* sWbF  = sWaF + 128 * KPAD;                     // 8*4  frags = 16384
    unsigned short* sMsgF = sWbF + 128 * 128;                      // 4*9  frags = 18432
    unsigned short* sMidF = sMsgF + TILE_E * KPAD;                 // 4*4  frags =  8192
    __shared__ int sSrc[TILE_E];
    __shared__ int sDst[TILE_E];

    const int tid  = threadIdx.x;       // 256
    const int lane = tid & 31;
    const int wave = tid >> 5;
    const int mi   = wave & 3;          // M tile index (16 edges each)
    const int ng   = wave >> 2;         // N half (4 tiles of 16 cols)

    // ---- stage bf16 weights in B-fragment layout, once per block ----
    for (int i = tid; i < MSG_K * HID; i += 256) {       // coalesced over n
        const int k = i >> 7, n = i & 127;
        sWaF[b_slot(k, n, NKT1)] = f2bf(Wa[i]);
    }
    for (int i = tid; i < (KPAD - MSG_K) * HID; i += 256) {
        const int k = MSG_K + (i >> 7), n = i & 127;
        sWaF[b_slot(k, n, NKT1)] = 0;
    }
    for (int i = tid; i < HID * HID; i += 256) {
        const int k = i >> 7, n = i & 127;
        sWbF[b_slot(k, n, NKT2)] = f2bf(Wb[i]);
    }
    __syncthreads();

    const long nTiles = ((long)nEdges + TILE_E - 1) / TILE_E;
    for (long tile = blockIdx.x; tile < nTiles; tile += gridDim.x) {
        const long e0 = tile * TILE_E;

        // ---- stage tile edge indices in LDS (clamp tail: max idempotent) ----
        if (tid < TILE_E) {
            long ge = e0 + tid; if (ge > nEdges - 1) ge = nEdges - 1;
            sSrc[tid] = src[ge];
            sDst[tid] = dst[ge];
            // prefetch next tile's edge-index lines (global_prefetch_b8)
            const long pe = ge + (long)gridDim.x * TILE_E;
            if (pe < nEdges) {
                __builtin_prefetch(&src[pe], 0, 0);
                __builtin_prefetch(&dst[pe], 0, 0);
            }
        }
        __syncthreads();

        // ---- build message tile directly in A-fragment layout ----
        for (int i = tid; i < TILE_E * HID; i += 256) {
            const int e = i >> 7, c = i & 127;
            const int s = sSrc[e], d = sDst[e];
            const float hj = hIn[(size_t)s * hStride + c];
            const float hi = hIn[(size_t)d * hStride + c];
            sMsgF[a_slot(e, c, NKT1)]       = f2bf_rz(hj - hi);
            sMsgF[a_slot(e, c + HID, NKT1)] = f2bf_rz(hi);
            if (c == 0) {
                const float dk = x[(size_t)s * 16] - x[(size_t)d * 16];
                sMsgF[a_slot(e, 2 * HID, NKT1)] = f2bf_rz(dk);
            } else if (c < KPAD - MSG_K + 1) {   // c in [1,31] zero-pads k=257..287
                sMsgF[a_slot(e, 2 * HID + c, NKT1)] = 0;
            }
        }
        __syncthreads();

        // ---- GEMM1: bf16 WMMA, K = 288 ----
        v8f acc[4];
#pragma unroll
        for (int t = 0; t < 4; ++t) acc[t] = (v8f){0.f,0.f,0.f,0.f,0.f,0.f,0.f,0.f};
#pragma unroll 1
        for (int kt = 0; kt < NKT1; ++kt) {
            const v16bf a = load_frag(sMsgF, mi * NKT1 + kt, lane);
#pragma unroll
            for (int t = 0; t < 4; ++t) {
                const v16bf b = load_frag(sWaF, (ng * 4 + t) * NKT1 + kt, lane);
                acc[t] = __builtin_amdgcn_wmma_f32_16x16x32_bf16(
                    false, a, false, b, (short)0, acc[t], false, false);
            }
        }

        // ---- bias + leaky relu -> A-fragment staging for GEMM2 ----
        const int hh = lane >> 4, nn = lane & 15;
#pragma unroll
        for (int t = 0; t < 4; ++t) {
            const int ncol = (ng * 4 + t) * 16 + nn;
            const float bias = ba[ncol];
#pragma unroll
            for (int r = 0; r < 8; ++r) {
                float v = acc[t][r] + bias;
                v = v > 0.f ? v : 0.1f * v;
                sMidF[a_slot(mi * 16 + r + 8 * hh, ncol, NKT2)] = f2bf_rz(v);
            }
        }
        __syncthreads();   // all columns of sMid ready

        // ---- GEMM2: bf16 WMMA, K = 128 ----
        v8f acc2[4];
#pragma unroll
        for (int t = 0; t < 4; ++t) acc2[t] = (v8f){0.f,0.f,0.f,0.f,0.f,0.f,0.f,0.f};
#pragma unroll
        for (int kt = 0; kt < NKT2; ++kt) {
            const v16bf a = load_frag(sMidF, mi * NKT2 + kt, lane);
#pragma unroll
            for (int t = 0; t < 4; ++t) {
                const v16bf b = load_frag(sWbF, (ng * 4 + t) * NKT2 + kt, lane);
                acc2[t] = __builtin_amdgcn_wmma_f32_16x16x32_bf16(
                    false, a, false, b, (short)0, acc2[t], false, false);
            }
        }

        // ---- bias + branchless key-encoded segment-max scatter ----
#pragma unroll
        for (int t = 0; t < 4; ++t) {
            const int ncol = (ng * 4 + t) * 16 + nn;
            const float bias = bb[ncol];
#pragma unroll
            for (int r = 0; r < 8; ++r) {
                const int e = mi * 16 + r + 8 * hh;
                const int d = sDst[e];
                atomicMax(&agg[(size_t)d * HID + ncol], f32key(acc2[t][r] + bias));
            }
        }
        __syncthreads();   // before sMsgF/sDst are overwritten by next tile
    }
}

// ---------------------------------------------------------------------------
// Kernel: layer finalize  out = relu( (isfinite(decode(agg))?...:0) + hIn )
// Untouched cells (key 0) decode to NaN -> become 0, matching the reference.
// ---------------------------------------------------------------------------
__global__ void finalize_kernel(const unsigned int* __restrict__ agg,
                                const float* __restrict__ hIn, long hStride,
                                float* __restrict__ out, long outStride) {
    long i = (long)blockIdx.x * blockDim.x + threadIdx.x;
    long stride = (long)gridDim.x * blockDim.x;
    const long total = (long)N_NODES * HID;
    for (; i < total; i += stride) {
        const long n = i >> 7, c = i & 127;
        float v = keyf32(agg[i]);
        if (!__builtin_isfinite(v)) v = 0.f;
        v += hIn[n * hStride + c];
        out[n * outStride + c] = v > 0.f ? v : 0.f;
    }
}

// ---------------------------------------------------------------------------
// Kernel: graph pooling (key-encoded atomic max + sum + count)
// ---------------------------------------------------------------------------
__global__ void pool_kernel(const float* __restrict__ cat,
                            const int* __restrict__ batch,
                            unsigned int* __restrict__ maxp,  // [G][256] keys, 0 init
                            float* __restrict__ sump,         // [G][256], 0 init
                            float* __restrict__ cnt) {        // [G], 0 init
    const int n = blockIdx.x;
    const int c = threadIdx.x;      // 256
    const int g = batch[n];
    const float v = cat[(size_t)n * 256 + c];
    atomicMax(&maxp[(size_t)g * 256 + c], f32key(v));
    atomicAdd(&sump[(size_t)g * 256 + c], v);
    if (c == 0) atomicAdd(&cnt[g], 1.0f);
}

// ---------------------------------------------------------------------------
// Kernel: head MLP + log_softmax, one block (128 threads) per graph
// ---------------------------------------------------------------------------
__global__ void head_kernel(const unsigned int* __restrict__ maxp,
                            const float* __restrict__ sump,
                            const float* __restrict__ cnt,
                            const float* __restrict__ Wf1,  // [512][128]
                            const float* __restrict__ bf1,
                            const float* __restrict__ Wf2,  // [128][2]
                            const float* __restrict__ bf2,
                            float* __restrict__ out) {      // [G][2]
    __shared__ float pooled[512];
    __shared__ float hid[128];
    __shared__ float logits[2];
    const int g = blockIdx.x;
    const int t = threadIdx.x;      // 128
    for (int i = t; i < 256; i += 128) {
        const float m = keyf32(maxp[(size_t)g * 256 + i]);
        pooled[i] = __builtin_isfinite(m) ? m : 0.f;
    }
    float c = cnt[g]; if (c < 1.f) c = 1.f;
    for (int i = t; i < 256; i += 128)
        pooled[256 + i] = sump[(size_t)g * 256 + i] / c;
    __syncthreads();
    float a = bf1[t];
#pragma unroll 1
    for (int k = 0; k < 512; ++k) a += pooled[k] * Wf1[k * HID + t];
    hid[t] = a > 0.f ? a : 0.f;
    __syncthreads();
    if (t < 2) {
        float o = bf2[t];
#pragma unroll 1
        for (int k = 0; k < 128; ++k) o += hid[k] * Wf2[k * 2 + t];
        logits[t] = o;
    }
    __syncthreads();
    if (t == 0) {
        const float l0 = logits[0], l1 = logits[1];
        const float m = l0 > l1 ? l0 : l1;
        const float lse = m + __logf(__expf(l0 - m) + __expf(l1 - m));
        out[g * 2 + 0] = l0 - lse;
        out[g * 2 + 1] = l1 - lse;
    }
}

// ---------------------------------------------------------------------------
// Host launcher
// ---------------------------------------------------------------------------
extern "C" void kernel_launch(void* const* d_in, const int* in_sizes, int n_in,
                              void* d_out, int out_size, void* d_ws, size_t ws_size,
                              hipStream_t stream) {
    const float* x     = (const float*)d_in[0];
    const float* W_enc = (const float*)d_in[1];
    const float* b_enc = (const float*)d_in[2];
    const float* W0    = (const float*)d_in[3];
    const float* b0    = (const float*)d_in[4];
    const float* W1a   = (const float*)d_in[5];
    const float* b1a   = (const float*)d_in[6];
    const float* W1b   = (const float*)d_in[7];
    const float* b1b   = (const float*)d_in[8];
    const float* W2a   = (const float*)d_in[9];
    const float* b2a   = (const float*)d_in[10];
    const float* W2b   = (const float*)d_in[11];
    const float* b2b   = (const float*)d_in[12];
    const float* Wf1   = (const float*)d_in[13];
    const float* bf1   = (const float*)d_in[14];
    const float* Wf2   = (const float*)d_in[15];
    const float* bf2   = (const float*)d_in[16];
    const int* src1    = (const int*)d_in[17];
    const int* dst1    = (const int*)d_in[18];
    const int* src2    = (const int*)d_in[19];
    const int* dst2    = (const int*)d_in[20];
    const int* batch   = (const int*)d_in[21];
    float* out = (float*)d_out;

    // Workspace carving (floats): h | cat | agg | maxp | sump | cnt
    float* ws   = (float*)d_ws;
    float* hBuf = ws;                                     // N_NODES*128
    float* cat  = hBuf + (size_t)N_NODES * HID;           // N_NODES*256
    unsigned int* agg  = (unsigned int*)(cat + (size_t)N_NODES * 256);  // N_NODES*128 keys
    unsigned int* maxp = agg + (size_t)N_NODES * HID;     // 128*256 keys
    float* sump = (float*)(maxp + (size_t)N_GRAPHS * 256);// 128*256
    float* cnt  = sump + (size_t)N_GRAPHS * 256;          // 128

    const long aggN = (long)N_NODES * HID;
    const size_t edgeSmem =
        (size_t)(128 * KPAD + 128 * 128 + TILE_E * KPAD + TILE_E * HID) * sizeof(unsigned short);

    // 1. encoder
    encoder_kernel<<<N_NODES, 128, 0, stream>>>(x, W_enc, b_enc, W0, b0, hBuf);

    // 2. layer 1: agg keys = 0; edge conv; finalize -> cat[:, :128]
    fill_u32_kernel<<<2048, 256, 0, stream>>>(agg, aggN, 0u);
    edge_conv_kernel<<<EDGE_BLOCKS, 256, edgeSmem, stream>>>(
        hBuf, HID, x, src1, dst1, N_EDGES, W1a, b1a, W1b, b1b, agg);
    finalize_kernel<<<4096, 256, 0, stream>>>(agg, hBuf, HID, cat, 256);

    // 3. layer 2: reads x1 = cat[:, :128] (stride 256); finalize -> cat[:, 128:]
    fill_u32_kernel<<<2048, 256, 0, stream>>>(agg, aggN, 0u);
    edge_conv_kernel<<<EDGE_BLOCKS, 256, edgeSmem, stream>>>(
        cat, 256, x, src2, dst2, N_EDGES, W2a, b2a, W2b, b2b, agg);
    finalize_kernel<<<4096, 256, 0, stream>>>(agg, cat, 256, cat + HID, 256);

    // 4. pooling
    fill_u32_kernel<<<64, 256, 0, stream>>>(maxp, (long)N_GRAPHS * 256, 0u);
    fill_u32_kernel<<<64, 256, 0, stream>>>((unsigned int*)sump, (long)N_GRAPHS * 256, 0u);
    fill_u32_kernel<<<1, 128, 0, stream>>>((unsigned int*)cnt, N_GRAPHS, 0u);
    pool_kernel<<<N_NODES, 256, 0, stream>>>(cat, batch, maxp, sump, cnt);

    // 5. head MLP + log_softmax
    head_kernel<<<N_GRAPHS, 128, 0, stream>>>(maxp, sump, cnt, Wf1, bf1, Wf2, bf2, out);
}
